// NetVLAD_31516470018310
// MI455X (gfx1250) — matrix operand order repaired and checked by reading, not compiled, not run on hardware
//
#include <hip/hip_runtime.h>
#include <math.h>

// ---------------------------------------------------------------------------
// NetVLAD for MI455X (gfx1250), wave32, bf16 WMMA (v_wmma_f32_16x16x32_bf16)
// B=16, C=128, H*W=N=4096, K=64
// ---------------------------------------------------------------------------

#define BATCH  16
#define CDIM   128
#define NLOC   4096
#define KCL    64
#define NCHUNK 8          // depth split for the VLAD GEMM

typedef __attribute__((ext_vector_type(16))) __bf16        v16bf;
typedef __attribute__((ext_vector_type(8)))  float         v8f;
typedef __attribute__((ext_vector_type(4)))  unsigned int  u32x4;

union Frag16 { v16bf v; u32x4 q[2]; };

__device__ __forceinline__ unsigned short f2bf(float f) {
  union { float f; unsigned u; } x; x.f = f;
  unsigned u = x.u;
  // round-to-nearest-even truncation to bf16
  return (unsigned short)((u + 0x7FFFu + ((u >> 16) & 1u)) >> 16);
}
__device__ __forceinline__ float bf2f(unsigned short b) {
  union { unsigned u; float f; } x; x.u = ((unsigned)b) << 16;
  return x.f;
}

// ---------------------------------------------------------------------------
// 1) channel-wise L2 normalize, emit bf16 in two layouts:
//    xnT [B, N, C]  (GEMM1 B operand: contiguous in C)
//    xnN [B, C, N]  (GEMM2 B operand: contiguous in N)
// ---------------------------------------------------------------------------
__global__ void k_normalize(const float* __restrict__ x,
                            unsigned short* __restrict__ xnN,
                            unsigned short* __restrict__ xnT) {
  int t = blockIdx.x * blockDim.x + threadIdx.x;   // 65536 = B*N
  int b = t >> 12, n = t & (NLOC - 1);
  const float* xb = x + (size_t)b * CDIM * NLOC + n;
  float ssq = 0.f;
  #pragma unroll 4
  for (int c = 0; c < CDIM; ++c) { float v = xb[(size_t)c * NLOC]; ssq += v * v; }
  float sc = 1.0f / fmaxf(sqrtf(ssq), 1e-12f);
  unsigned short* oN = xnN + (size_t)b * CDIM * NLOC + n;
  unsigned short* oT = xnT + ((size_t)b * NLOC + n) * CDIM;
  #pragma unroll 4
  for (int c = 0; c < CDIM; ++c) {
    unsigned short bb = f2bf(xb[(size_t)c * NLOC] * sc);
    oN[(size_t)c * NLOC] = bb;
    oT[c] = bb;
  }
}

// conv_w fp32 -> bf16
__global__ void k_transcode_w(const float* __restrict__ w,
                              unsigned short* __restrict__ cwb) {
  int i = blockIdx.x * blockDim.x + threadIdx.x;   // 8192
  cwb[i] = f2bf(w[i]);
}

// ---------------------------------------------------------------------------
// 2) assignment: logits = conv_w @ xn + b, softmax over K=64, a -> bf16
//    one wave per 16-column n-tile; 4 M-tiles x 4 depth steps = 16 WMMAs
// ---------------------------------------------------------------------------
__global__ void k_assign(const unsigned short* __restrict__ cwb,    // [K,C] bf16
                         const float*          __restrict__ conv_b, // [K]
                         const unsigned short* __restrict__ xnT,    // [B,N,C] bf16
                         unsigned short*       __restrict__ a) {    // [B,K,N] bf16
  __shared__ float sb[KCL];
  if (threadIdx.x < KCL) sb[threadIdx.x] = conv_b[threadIdx.x];
  __syncthreads();

  int wid  = threadIdx.x >> 5;
  int lane = threadIdx.x & 31;
  int tile = blockIdx.x * 4 + wid;          // B * N/16 = 4096 tiles
  int b      = tile >> 8;
  int nblock = (tile & 255) << 4;
  int l15 = lane & 15;
  int hi  = lane >> 4;

  // init accumulators with the bias (C operand of first WMMA step)
  v8f acc[4];
  #pragma unroll
  for (int mt = 0; mt < 4; ++mt)
    #pragma unroll
    for (int r = 0; r < 8; ++r)
      acc[mt][r] = sb[16 * mt + r + 8 * hi];

  const unsigned short* xrow =
      xnT + ((size_t)b * NLOC + nblock + l15) * CDIM + hi * 8;

  #pragma unroll
  for (int cb = 0; cb < CDIM; cb += 32) {
    Frag16 Bf;                                       // B: 32(C) x 16(n)
    Bf.q[0] = *(const u32x4*)(xrow + cb);
    Bf.q[1] = *(const u32x4*)(xrow + cb + 16);
    #pragma unroll
    for (int mt = 0; mt < 4; ++mt) {                 // A: 16(k) x 32(C)
      const unsigned short* wrow =
          cwb + (size_t)(16 * mt + l15) * CDIM + cb + hi * 8;
      Frag16 Af;
      Af.q[0] = *(const u32x4*)(wrow);
      Af.q[1] = *(const u32x4*)(wrow + 16);
      acc[mt] = __builtin_amdgcn_wmma_f32_16x16x32_bf16(
          false, Af.v, false, Bf.v, (short)0, acc[mt], false, false);
    }
  }

  // softmax over K=64 per column: lanes L and L^16 split the column
  float mx = -3.402823466e38f;
  #pragma unroll
  for (int mt = 0; mt < 4; ++mt)
    #pragma unroll
    for (int r = 0; r < 8; ++r) mx = fmaxf(mx, acc[mt][r]);
  mx = fmaxf(mx, __shfl_xor(mx, 16, 32));

  float sum = 0.f;
  #pragma unroll
  for (int mt = 0; mt < 4; ++mt)
    #pragma unroll
    for (int r = 0; r < 8; ++r) {
      float e = expf(acc[mt][r] - mx);
      acc[mt][r] = e;
      sum += e;
    }
  sum += __shfl_xor(sum, 16, 32);
  float inv = 1.0f / sum;

  #pragma unroll
  for (int mt = 0; mt < 4; ++mt)
    #pragma unroll
    for (int r = 0; r < 8; ++r) {
      int k = 16 * mt + r + 8 * hi;
      a[((size_t)b * KCL + k) * NLOC + nblock + l15] = f2bf(acc[mt][r] * inv);
    }
}

// ---------------------------------------------------------------------------
// 3) asum[b,k] = sum_n a[b,k,n]
// ---------------------------------------------------------------------------
__global__ void k_asum(const unsigned short* __restrict__ a,
                       float* __restrict__ asum) {
  int bk = blockIdx.x;                         // 1024
  const unsigned short* row = a + (size_t)bk * NLOC;
  float s = 0.f;
  for (int i = threadIdx.x; i < NLOC; i += blockDim.x) s += bf2f(row[i]);
  __shared__ float red[128];
  red[threadIdx.x] = s;
  __syncthreads();
  for (int off = 64; off > 0; off >>= 1) {
    if (threadIdx.x < off) red[threadIdx.x] += red[threadIdx.x + off];
    __syncthreads();
  }
  if (threadIdx.x == 0) asum[bk] = red[0];
}

// ---------------------------------------------------------------------------
// 4) VLAD GEMM: vlad[b,k,c] += sum_n a[b,k,n]*xn[b,c,n]
//    one wave = one (b, k-tile, depth-chunk); 8 C-tiles in registers
// ---------------------------------------------------------------------------
__global__ void k_vlad(const unsigned short* __restrict__ a,    // [B,K,N] bf16
                       const unsigned short* __restrict__ xnN,  // [B,C,N] bf16
                       float* __restrict__ vlad) {              // [B,K,C] f32
  int wid  = threadIdx.x >> 5;
  int lane = threadIdx.x & 31;
  int w = blockIdx.x * 4 + wid;            // B*4*NCHUNK = 512 waves
  int b     = w / (4 * NCHUNK);
  int rem   = w % (4 * NCHUNK);
  int mt    = rem / NCHUNK;
  int chunk = rem % NCHUNK;
  int l15 = lane & 15;
  int hi  = lane >> 4;

  v8f acc[8] = {};

  const unsigned short* arow =
      a + ((size_t)b * KCL + 16 * mt + l15) * NLOC + hi * 8;
  const unsigned short* xbase =
      xnN + (size_t)b * CDIM * NLOC + hi * 8;

  int n0 = chunk * (NLOC / NCHUNK);
  int n1 = n0 + (NLOC / NCHUNK);
  for (int nb = n0; nb < n1; nb += 32) {
    Frag16 Af;                                  // A: 16(k) x 32(n)
    Af.q[0] = *(const u32x4*)(arow + nb);
    Af.q[1] = *(const u32x4*)(arow + nb + 16);
    #pragma unroll
    for (int ct = 0; ct < 8; ++ct) {            // B: 32(n) x 16(c)
      const unsigned short* xr = xbase + (size_t)(ct * 16 + l15) * NLOC + nb;
      Frag16 Bf;
      Bf.q[0] = *(const u32x4*)(xr);
      Bf.q[1] = *(const u32x4*)(xr + 16);
      acc[ct] = __builtin_amdgcn_wmma_f32_16x16x32_bf16(
          false, Af.v, false, Bf.v, (short)0, acc[ct], false, false);
    }
  }

  #pragma unroll
  for (int ct = 0; ct < 8; ++ct)
    #pragma unroll
    for (int r = 0; r < 8; ++r) {
      int k = 16 * mt + r + 8 * hi;
      int c = ct * 16 + l15;
      atomicAdd(&vlad[((size_t)b * KCL + k) * CDIM + c], acc[ct][r]);
    }
}

// ---------------------------------------------------------------------------
// 5) subtract centroids*asum, intra-normalize over C, accumulate batch ssq
// ---------------------------------------------------------------------------
__global__ void k_intranorm(const float* __restrict__ vlad,
                            const float* __restrict__ centroids,
                            const float* __restrict__ asum,
                            float* __restrict__ out,
                            float* __restrict__ bssq) {
  int wid  = threadIdx.x >> 5;
  int lane = threadIdx.x & 31;
  int bk = blockIdx.x * 4 + wid;       // 1024 rows
  int b = bk >> 6, k = bk & 63;
  float as = asum[bk];

  float v[4];
  float ssq = 0.f;
  #pragma unroll
  for (int i = 0; i < 4; ++i) {
    int c = lane + 32 * i;
    float t = vlad[(size_t)bk * CDIM + c] - centroids[k * CDIM + c] * as;
    v[i] = t;
    ssq += t * t;
  }
  #pragma unroll
  for (int m = 16; m >= 1; m >>= 1) ssq += __shfl_xor(ssq, m, 32);
  float rn = 1.0f / fmaxf(sqrtf(ssq), 1e-12f);

  float osq = 0.f;
  #pragma unroll
  for (int i = 0; i < 4; ++i) {
    int c = lane + 32 * i;
    float o = v[i] * rn;
    out[(size_t)bk * CDIM + c] = o;
    osq += o * o;
  }
  #pragma unroll
  for (int m = 16; m >= 1; m >>= 1) osq += __shfl_xor(osq, m, 32);
  if (lane == 0) atomicAdd(&bssq[b], osq);
}

// ---------------------------------------------------------------------------
// 6) global per-batch L2 normalize
// ---------------------------------------------------------------------------
__global__ void k_globalnorm(float* __restrict__ out,
                             const float* __restrict__ bssq) {
  int i = blockIdx.x * blockDim.x + threadIdx.x;  // 131072
  int b = i >> 13;
  out[i] *= 1.0f / fmaxf(sqrtf(bssq[b]), 1e-12f);
}

// ---------------------------------------------------------------------------
extern "C" void kernel_launch(void* const* d_in, const int* in_sizes, int n_in,
                              void* d_out, int out_size, void* d_ws, size_t ws_size,
                              hipStream_t stream) {
  (void)in_sizes; (void)n_in; (void)out_size; (void)ws_size;
  const float* x         = (const float*)d_in[0];   // [16,128,64,64]
  const float* conv_w    = (const float*)d_in[1];   // [64,128]
  const float* conv_b    = (const float*)d_in[2];   // [64]
  const float* centroids = (const float*)d_in[3];   // [64,128]
  float* out = (float*)d_out;                       // [16, 8192]

  char* ws = (char*)d_ws;
  unsigned short* xnT = (unsigned short*)(ws);              // 16 MiB: [B,N,C] bf16
  unsigned short* xnN = (unsigned short*)(ws + 16777216);   // 16 MiB: [B,C,N] bf16
  unsigned short* a_  = (unsigned short*)(ws + 33554432);   //  8 MiB: [B,K,N] bf16
  unsigned short* cwb = (unsigned short*)(ws + 41943040);   // 16 KiB: [K,C] bf16
  float* asum = (float*)(ws + 41959424);                    //  4 KiB: [B*K]
  float* vlad = (float*)(ws + 41963520);                    // 512 KiB: [B,K,C] f32
  float* bssq = (float*)(ws + 42487808);                    //  64 B : [B]

  // atomic accumulators must start at zero every launch (graph-capturable)
  hipMemsetAsync(vlad, 0, (size_t)BATCH * KCL * CDIM * sizeof(float), stream);
  hipMemsetAsync(bssq, 0, (size_t)BATCH * sizeof(float), stream);

  k_transcode_w<<<32, 256, 0, stream>>>(conv_w, cwb);
  k_normalize<<<(BATCH * NLOC) / 256, 256, 0, stream>>>(x, xnN, xnT);
  k_assign<<<(BATCH * (NLOC / 16)) / 4, 128, 0, stream>>>(cwb, conv_b, xnT, a_);
  k_asum<<<BATCH * KCL, 128, 0, stream>>>(a_, asum);
  k_vlad<<<(BATCH * 4 * NCHUNK) / 4, 128, 0, stream>>>(a_, xnN, vlad);
  k_intranorm<<<(BATCH * KCL) / 4, 128, 0, stream>>>(vlad, centroids, asum, out, bssq);
  k_globalnorm<<<(BATCH * KCL * CDIM) / 256, 256, 0, stream>>>(out, bssq);
}
